// BigMacMoE_25005299598049
// MI455X (gfx1250) — matrix operand order repaired and checked by reading, not compiled, not run on hardware
//
#include <hip/hip_runtime.h>
#include <hip/hip_bf16.h>

typedef __attribute__((ext_vector_type(16))) __bf16 v16bf;
typedef __attribute__((ext_vector_type(8)))  float  v8f;

#define NTOK      4096
#define DMODEL    2048
#define NEXP      64
#define TOPK      4
#define NKTOT     (NTOK*TOPK)
#define MAXLOAD   304
#define NUM_SLOTS (NEXP*MAXLOAD)
#define DFF       8192
#define BN        512
#define EHALF     1024

static __device__ __forceinline__ float sigmoidf_(float v){ return 1.0f/(1.0f+__expf(-v)); }
static __device__ __forceinline__ float siluf_(float v){ return v * sigmoidf_(v); }

static __device__ __forceinline__ v16bf pack16_(const float* t){
  v16bf v;
#pragma unroll
  for (int i=0;i<16;i++) v[i] = (__bf16)t[i];
  return v;
}

static __device__ __forceinline__ v8f wmma_bf16_(v16bf a, v16bf b, v8f c){
  return __builtin_amdgcn_wmma_f32_16x16x32_bf16(false, a, false, b, (short)0, c, false, false);
}

// A fragment: 16x32 slice of row-major f32 A. Contiguous K layout:
// lane<16 holds M=lane, K=[0..15]; lane>=16 holds M=lane-16, K=[16..31].
static __device__ __forceinline__ v16bf load_a_f32_(const float* __restrict__ A, int lda,
                                                    int m0, int k0, int lane){
  const int lr = lane & 15, hf = lane >> 4;
  const float* p = A + (size_t)(m0 + lr) * lda + k0 + hf * 16;
  float t[16];
  float4 c0 = *(const float4*)(p);
  float4 c1 = *(const float4*)(p + 4);
  float4 c2 = *(const float4*)(p + 8);
  float4 c3 = *(const float4*)(p + 12);
  t[0]=c0.x; t[1]=c0.y; t[2]=c0.z; t[3]=c0.w;
  t[4]=c1.x; t[5]=c1.y; t[6]=c1.z; t[7]=c1.w;
  t[8]=c2.x; t[9]=c2.y; t[10]=c2.z; t[11]=c2.w;
  t[12]=c3.x; t[13]=c3.y; t[14]=c3.z; t[15]=c3.w;
  return pack16_(t);
}

// Same as above but rows indirected through a slot->token map (sentinel row >= NTOK -> zeros).
static __device__ __forceinline__ v16bf load_a_gather_(const float* __restrict__ A, int lda,
                                                       const int* __restrict__ rmap,
                                                       int m0, int k0, int lane){
  const int lr = lane & 15, hf = lane >> 4;
  const int row = rmap[m0 + lr];
  float t[16];
  if (row < NTOK){
    const float* p = A + (size_t)row * lda + k0 + hf * 16;
    float4 c0 = *(const float4*)(p);
    float4 c1 = *(const float4*)(p + 4);
    float4 c2 = *(const float4*)(p + 8);
    float4 c3 = *(const float4*)(p + 12);
    t[0]=c0.x; t[1]=c0.y; t[2]=c0.z; t[3]=c0.w;
    t[4]=c1.x; t[5]=c1.y; t[6]=c1.z; t[7]=c1.w;
    t[8]=c2.x; t[9]=c2.y; t[10]=c2.z; t[11]=c2.w;
    t[12]=c3.x; t[13]=c3.y; t[14]=c3.z; t[15]=c3.w;
  } else {
#pragma unroll
    for (int i=0;i<16;i++) t[i]=0.0f;
  }
  return pack16_(t);
}

// B fragment, weight stored (N,K) row-major (i.e. C = A @ W^T). Lane = output column,
// 16 consecutive K values -> fully contiguous 64B load per lane.
static __device__ __forceinline__ v16bf load_b_nk_(const float* __restrict__ W, int ldw,
                                                   int n0, int k0, int lane){
  const int lr = lane & 15, hf = lane >> 4;
  const float* p = W + (size_t)(n0 + lr) * ldw + k0 + hf * 16;
  float t[16];
  float4 c0 = *(const float4*)(p);
  float4 c1 = *(const float4*)(p + 4);
  float4 c2 = *(const float4*)(p + 8);
  float4 c3 = *(const float4*)(p + 12);
  t[0]=c0.x; t[1]=c0.y; t[2]=c0.z; t[3]=c0.w;
  t[4]=c1.x; t[5]=c1.y; t[6]=c1.z; t[7]=c1.w;
  t[8]=c2.x; t[9]=c2.y; t[10]=c2.z; t[11]=c2.w;
  t[12]=c3.x; t[13]=c3.y; t[14]=c3.z; t[15]=c3.w;
  return pack16_(t);
}

// B fragment, weight stored (K,N) row-major (expert weights): strided along K,
// coalesced across lanes at fixed K.
static __device__ __forceinline__ v16bf load_b_kn_(const float* __restrict__ B, int ldb,
                                                   int n0, int k0, int lane){
  const int lr = lane & 15, hf = lane >> 4;
  const float* p = B + (size_t)(k0 + hf * 16) * ldb + n0 + lr;
  float t[16];
#pragma unroll
  for (int i=0;i<16;i++) t[i] = p[(size_t)i * ldb];
  return pack16_(t);
}

// ---------------------------------------------------------------------------
// Kernel 0: zero the z-loss accumulator slot.
__global__ void zero_zloss_kernel(float* z){ if (threadIdx.x==0 && blockIdx.x==0) z[0]=0.0f; }

// ---------------------------------------------------------------------------
// Kernel 1: router logits, z-loss, sigmoid+bias top-4, softmax gating.
__global__ void router_topk_kernel(const float* __restrict__ x,
                                   const float* __restrict__ router_w,
                                   const float* __restrict__ ebias,
                                   int* __restrict__ topk_idx,
                                   float* __restrict__ gating,
                                   float* __restrict__ zloss){
  __shared__ float logits_s[NEXP];
  __shared__ float sel_s[NEXP];
  __shared__ float sq_s[NEXP];
  const int token = blockIdx.x;
  const int e = threadIdx.x;              // 64 threads
  const float4* hv = (const float4*)(x + (size_t)token * DMODEL);
  const float4* wv = (const float4*)(router_w + (size_t)e * DMODEL);
  float acc = 0.0f;
  for (int i=0;i<DMODEL/4;i++){
    float4 a = hv[i], b = wv[i];
    acc += a.x*b.x + a.y*b.y + a.z*b.z + a.w*b.w;
  }
  logits_s[e] = acc;
  sq_s[e]     = acc*acc;
  sel_s[e]    = sigmoidf_(acc) + ebias[e];
  __syncthreads();
  if (e == 0){
    float zs = 0.0f;
    for (int j=0;j<NEXP;j++) zs += sq_s[j];
    atomicAdd(zloss, zs * (1e-4f / ((float)NTOK * (float)NEXP)));
    unsigned long long used = 0ull;
    int   idxk[TOPK];
    float logk[TOPK];
    for (int k=0;k<TOPK;k++){
      float best = -1e30f; int bi = 0;
      for (int j=0;j<NEXP;j++){
        if (used & (1ull<<j)) continue;
        float s = sel_s[j];
        if (s > best){ best = s; bi = j; }
      }
      used |= (1ull<<bi);
      idxk[k] = bi; logk[k] = logits_s[bi];
    }
    float mx = fmaxf(fmaxf(logk[0],logk[1]), fmaxf(logk[2],logk[3]));
    float ex[TOPK]; float ssum = 0.0f;
    for (int k=0;k<TOPK;k++){ ex[k] = __expf(logk[k]-mx); ssum += ex[k]; }
    for (int k=0;k<TOPK;k++){
      topk_idx[token*TOPK+k] = idxk[k];
      gating  [token*TOPK+k] = ex[k]/ssum;
    }
  }
}

// ---------------------------------------------------------------------------
// Kernel 2: stable per-expert ranking -> reverse_map (slot->token) and gather_map.
__global__ void routing_kernel(const int* __restrict__ flat,
                               int* __restrict__ rmap,
                               int* __restrict__ gmap){
  const int e = blockIdx.x;               // one block per expert
  const int t = threadIdx.x;              // 256 threads
  const int CH = NKTOT / 256;             // 64 entries each
  __shared__ int scnt[256];
  __shared__ int soff[257];
  const int base = t * CH;
  int c = 0;
  for (int i=0;i<CH;i++) c += (flat[base+i] == e) ? 1 : 0;
  scnt[t] = c;
  __syncthreads();
  if (t == 0){
    int run = 0;
    for (int j=0;j<256;j++){ soff[j] = run; run += scnt[j]; }
    soff[256] = run;
  }
  __syncthreads();
  int run = soff[t];
  for (int i=0;i<CH;i++){
    int idx = base + i;
    if (flat[idx] == e){
      int r = run++;
      if (r < MAXLOAD){
        rmap[e*MAXLOAD + r] = idx >> 2;         // token = idx / TOPK
        gmap[idx] = e*MAXLOAD + r;
      } else {
        gmap[idx] = NUM_SLOTS;                  // dropped -> zero row
      }
    }
  }
  const int total = soff[256];
  for (int j = total + t; j < MAXLOAD; j += 256) rmap[e*MAXLOAD + j] = NTOK; // sentinel
}

// ---------------------------------------------------------------------------
// Kernel 3: fused gate/up: mid = silu(h@gate_w^T) * (h@up_w^T).  M=4096 N=8192 K=2048.
// Wave tile: 32x32, dual-matrix (8 f32 accumulators, 8 WMMAs per K-step).
__global__ void gemm_gateup_kernel(const float* __restrict__ h,
                                   const float* __restrict__ gate_w,
                                   const float* __restrict__ up_w,
                                   float* __restrict__ mid){
  const int wave = threadIdx.x >> 5;
  const int lane = threadIdx.x & 31;
  const int n0 = (blockIdx.x * 8 + wave) * 32;
  const int m0 = blockIdx.y * 32;
  v8f g00={},g01={},g10={},g11={};
  v8f u00={},u01={},u10={},u11={};
  for (int k0=0;k0<DMODEL;k0+=32){
    __builtin_prefetch(gate_w + (size_t)n0*DMODEL + k0 + 64, 0, 1);
    v16bf a0  = load_a_f32_(h, DMODEL, m0,    k0, lane);
    v16bf a1  = load_a_f32_(h, DMODEL, m0+16, k0, lane);
    v16bf bg0 = load_b_nk_(gate_w, DMODEL, n0,    k0, lane);
    v16bf bg1 = load_b_nk_(gate_w, DMODEL, n0+16, k0, lane);
    v16bf bu0 = load_b_nk_(up_w,   DMODEL, n0,    k0, lane);
    v16bf bu1 = load_b_nk_(up_w,   DMODEL, n0+16, k0, lane);
    g00 = wmma_bf16_(a0,bg0,g00);  g01 = wmma_bf16_(a0,bg1,g01);
    g10 = wmma_bf16_(a1,bg0,g10);  g11 = wmma_bf16_(a1,bg1,g11);
    u00 = wmma_bf16_(a0,bu0,u00);  u01 = wmma_bf16_(a0,bu1,u01);
    u10 = wmma_bf16_(a1,bu0,u10);  u11 = wmma_bf16_(a1,bu1,u11);
  }
  const int lr = lane & 15, hf = lane >> 4;
#pragma unroll
  for (int r=0;r<8;r++){
    int row0 = m0 + r + hf*8, row1 = row0 + 16;
    int col0 = n0 + lr,       col1 = col0 + 16;
    mid[(size_t)row0*DFF + col0] = siluf_(g00[r]) * u00[r];
    mid[(size_t)row0*DFF + col1] = siluf_(g01[r]) * u01[r];
    mid[(size_t)row1*DFF + col0] = siluf_(g10[r]) * u10[r];
    mid[(size_t)row1*DFF + col1] = siluf_(g11[r]) * u11[r];
  }
}

// ---------------------------------------------------------------------------
// Kernel 4: generic C = A @ W^T (W stored (N,K)), 32x32 wave tile.
// epi==1: C = (addsrc + C) * 0.5
__global__ void wmma_gemm_nk_kernel(const float* __restrict__ A, int lda,
                                    const float* __restrict__ W, int ldw,
                                    float* __restrict__ C, int ldc,
                                    int K, int epi,
                                    const float* __restrict__ addsrc, int ldadd){
  const int wave = threadIdx.x >> 5;
  const int lane = threadIdx.x & 31;
  const int n0 = (blockIdx.x * 8 + wave) * 32;
  const int m0 = blockIdx.y * 32;
  v8f c00={},c01={},c10={},c11={};
  for (int k0=0;k0<K;k0+=32){
    v16bf a0 = load_a_f32_(A, lda, m0,    k0, lane);
    v16bf a1 = load_a_f32_(A, lda, m0+16, k0, lane);
    v16bf b0 = load_b_nk_(W, ldw, n0,    k0, lane);
    v16bf b1 = load_b_nk_(W, ldw, n0+16, k0, lane);
    c00 = wmma_bf16_(a0,b0,c00);  c01 = wmma_bf16_(a0,b1,c01);
    c10 = wmma_bf16_(a1,b0,c10);  c11 = wmma_bf16_(a1,b1,c11);
  }
  const int lr = lane & 15, hf = lane >> 4;
#pragma unroll
  for (int r=0;r<8;r++){
    int row0 = m0 + r + hf*8, row1 = row0 + 16;
    int col0 = n0 + lr,       col1 = col0 + 16;
    float v00=c00[r], v01=c01[r], v10=c10[r], v11=c11[r];
    if (epi == 1){
      v00 = (addsrc[(size_t)row0*ldadd + col0] + v00) * 0.5f;
      v01 = (addsrc[(size_t)row0*ldadd + col1] + v01) * 0.5f;
      v10 = (addsrc[(size_t)row1*ldadd + col0] + v10) * 0.5f;
      v11 = (addsrc[(size_t)row1*ldadd + col1] + v11) * 0.5f;
    }
    C[(size_t)row0*ldc + col0] = v00;
    C[(size_t)row0*ldc + col1] = v01;
    C[(size_t)row1*ldc + col0] = v10;
    C[(size_t)row1*ldc + col1] = v11;
  }
}

// ---------------------------------------------------------------------------
// Kernel 5: expert GEMM 1 with gather + fused silu-mul.
// act[e][m][n] = silu(h12[:,n]) * h12[:,n+1024],  h12 = padding @ w12[e]  (w12 is (K=512,N=2048))
__global__ void gemm_expert1_kernel(const float* __restrict__ hd,
                                    const float* __restrict__ ew12,
                                    const int* __restrict__ rmap,
                                    float* __restrict__ act){
  const int e = blockIdx.z;
  const int wave = threadIdx.x >> 5;
  const int lane = threadIdx.x & 31;
  const int n0 = (blockIdx.x * 8 + wave) * 16;   // in [0, 1024)
  const int m0 = blockIdx.y * 16;                // 19 tiles of 16 -> 304
  const int* rm = rmap + e * MAXLOAD;
  const float* B = ew12 + (size_t)e * BN * (2*EHALF);
  v8f c1={}, c2={};
  for (int k0=0;k0<BN;k0+=32){
    v16bf a  = load_a_gather_(hd, BN, rm, m0, k0, lane);
    v16bf b1 = load_b_kn_(B, 2*EHALF, n0,         k0, lane);
    v16bf b2 = load_b_kn_(B, 2*EHALF, n0 + EHALF, k0, lane);
    c1 = wmma_bf16_(a, b1, c1);
    c2 = wmma_bf16_(a, b2, c2);
  }
  const int lr = lane & 15, hf = lane >> 4;
  float* out = act + (size_t)e * MAXLOAD * EHALF;
#pragma unroll
  for (int r=0;r<8;r++){
    int row = m0 + r + hf*8;
    int col = n0 + lr;
    out[(size_t)row*EHALF + col] = siluf_(c1[r]) * c2[r];
  }
}

// ---------------------------------------------------------------------------
// Kernel 6: expert GEMM 2: pout[e] = act[e] (304x1024) @ w3[e] ((K=1024,N=512)).
__global__ void gemm_expert2_kernel(const float* __restrict__ act,
                                    const float* __restrict__ ew3,
                                    float* __restrict__ pout){
  const int e = blockIdx.z;
  const int wave = threadIdx.x >> 5;
  const int lane = threadIdx.x & 31;
  const int n0 = (blockIdx.x * 8 + wave) * 16;   // in [0, 512)
  const int m0 = blockIdx.y * 16;
  const float* A = act + (size_t)e * MAXLOAD * EHALF;
  const float* B = ew3 + (size_t)e * EHALF * BN;
  v8f c0={};
  for (int k0=0;k0<EHALF;k0+=32){
    v16bf a = load_a_f32_(A, EHALF, m0, k0, lane);
    v16bf b = load_b_kn_(B, BN, n0, k0, lane);
    c0 = wmma_bf16_(a, b, c0);
  }
  const int lr = lane & 15, hf = lane >> 4;
  float* out = pout + (size_t)e * MAXLOAD * BN;
#pragma unroll
  for (int r=0;r<8;r++){
    int row = m0 + r + hf*8;
    int col = n0 + lr;
    out[(size_t)row*BN + col] = c0[r];
  }
}

// ---------------------------------------------------------------------------
// Kernel 7: routed_b[n][b] = sum_k gating[n,k] * pout[gather_map[n*K+k]][b]
__global__ void gather_combine_kernel(const float* __restrict__ pout,
                                      const int* __restrict__ gmap,
                                      const float* __restrict__ gating,
                                      float* __restrict__ routed_b){
  const int idx = blockIdx.x * blockDim.x + threadIdx.x;  // 4096*512 threads
  const int n = idx >> 9;
  const int b = idx & (BN-1);
  float s = 0.0f;
#pragma unroll
  for (int k=0;k<TOPK;k++){
    int g = gmap[n*TOPK + k];
    if (g < NUM_SLOTS) s += gating[n*TOPK + k] * pout[(size_t)g*BN + b];
  }
  routed_b[idx] = s;
}

// ---------------------------------------------------------------------------
extern "C" void kernel_launch(void* const* d_in, const int* in_sizes, int n_in,
                              void* d_out, int out_size, void* d_ws, size_t ws_size,
                              hipStream_t stream){
  const float* x        = (const float*)d_in[0];
  const float* ebias    = (const float*)d_in[1];
  const float* router_w = (const float*)d_in[2];
  const float* gate_w   = (const float*)d_in[3];
  const float* up_w     = (const float*)d_in[4];
  const float* down_w   = (const float*)d_in[5];
  const float* w_down   = (const float*)d_in[6];
  const float* w_up     = (const float*)d_in[7];
  const float* ew12     = (const float*)d_in[8];
  const float* ew3      = (const float*)d_in[9];
  float* out   = (float*)d_out;
  float* zloss = out + (size_t)NTOK * DMODEL;

  char* ws = (char*)d_ws;
  int*   topk_idx = (int*)(ws + 0);                 //  64 KB
  float* gating   = (float*)(ws + 65536);           //  64 KB
  int*   rmap     = (int*)(ws + 131072);            //  ~76 KB
  int*   gmap     = (int*)(ws + 212992);            //  64 KB
  float* hd       = (float*)(ws + 278528);          //  8 MB
  float* shared_b = (float*)(ws + 8667136);         //  32 MB
  float* routed_b = (float*)(ws + 42221568);        //  8 MB
  float* mid      = (float*)(ws + 50610176);        //  128 MB
  float* act      = (float*)(ws + 50610176);        //  reuses mid after down-proj
  float* pout     = (float*)(ws + 130301952);       //  reuses mid tail

  zero_zloss_kernel<<<1, 1, 0, stream>>>(zloss);
  router_topk_kernel<<<NTOK, 64, 0, stream>>>(x, router_w, ebias, topk_idx, gating, zloss);
  // shared path
  gemm_gateup_kernel<<<dim3(DFF/256, NTOK/32), 256, 0, stream>>>(x, gate_w, up_w, mid);
  // hd = h @ w_down^T  (M=4096, N=512, K=2048)
  wmma_gemm_nk_kernel<<<dim3(BN/256, NTOK/32), 256, 0, stream>>>(
      x, DMODEL, w_down, DMODEL, hd, BN, DMODEL, 0, nullptr, 0);
  routing_kernel<<<NEXP, 256, 0, stream>>>(topk_idx, rmap, gmap);
  // shared = mid @ down_w^T  (M=4096, N=2048, K=8192)
  wmma_gemm_nk_kernel<<<dim3(DMODEL/256, NTOK/32), 256, 0, stream>>>(
      mid, DFF, down_w, DFF, shared_b, DMODEL, DFF, 0, nullptr, 0);
  // experts
  gemm_expert1_kernel<<<dim3(EHALF/128, MAXLOAD/16, NEXP), 256, 0, stream>>>(hd, ew12, rmap, act);
  gemm_expert2_kernel<<<dim3(BN/128, MAXLOAD/16, NEXP), 256, 0, stream>>>(act, ew3, pout);
  gather_combine_kernel<<<(NTOK*BN)/256, 256, 0, stream>>>(pout, gmap, gating, routed_b);
  // out = (shared + routed_b @ w_up^T) * 0.5  (M=4096, N=2048, K=512)
  wmma_gemm_nk_kernel<<<dim3(DMODEL/256, NTOK/32), 256, 0, stream>>>(
      routed_b, BN, w_up, BN, out, DMODEL, BN, 1, shared_b, DMODEL);
}
